// DeformConv2D_EPF_60868276519454
// MI455X (gfx1250) — matrix operand order repaired and checked by reading, not compiled, not run on hardware
//
#include <hip/hip_runtime.h>
#include <hip/hip_bf16.h>

// ---------------- problem constants ----------------
#define B_    512
#define C_    202
#define P_    27
#define CH_   200
#define N_    729          // P*P
#define K_    58           // int(N*0.08)
#define OUTC_ 64
#define CP_   224          // CH padded to multiple of 32
#define TAPS_ 9
#define KTOT_ (TAPS_*CP_)  // 2016
#define CHUNKS_ (KTOT_/32) // 63

typedef __bf16 v16bf __attribute__((ext_vector_type(16)));
typedef float  v8f   __attribute__((ext_vector_type(8)));

__device__ __forceinline__ unsigned short f2bf(float f) {
    unsigned int u = __float_as_uint(f);
    unsigned int r = u + 0x7FFFu + ((u >> 16) & 1u);   // round-to-nearest-even
    return (unsigned short)(r >> 16);
}

// ---------------------------------------------------------------------------
// Kernel 0: zero page used as async-copy source for padded/out-of-range lanes
// ---------------------------------------------------------------------------
__global__ void init_zero_kernel(unsigned int* __restrict__ zp) {
    if (threadIdx.x < 64) zp[threadIdx.x] = 0u;        // 256 B of zeros
}

// ---------------------------------------------------------------------------
// Kernel 1: per-batch EPF source index:  src[b][n] = n if masked else
//           masked_list[ zrank[n] % num_masked ]
// ---------------------------------------------------------------------------
__global__ void build_src_kernel(const float* __restrict__ x,
                                 const int* __restrict__ rand_idx,
                                 int* __restrict__ src) {
    const int b = blockIdx.x;
    __shared__ unsigned char  maskf[N_ + 7];
    __shared__ unsigned short list [N_ + 7];
    __shared__ unsigned short zr   [N_ + 7];
    __shared__ int num_one;

    const float* sp = x + ((size_t)b * C_ + CH_) * N_;   // channel 200
    const float central = sp[(P_/2) * P_ + (P_/2)];      // pixel 364

    for (int n = threadIdx.x; n < N_; n += blockDim.x)
        maskf[n] = (sp[n] == central) ? 1 : 0;
    __syncthreads();
    for (int i = threadIdx.x; i < K_; i += blockDim.x)
        maskf[rand_idx[b * K_ + i]] = 1;                 // forced positions
    __syncthreads();

    if (threadIdx.x == 0) {                              // tiny serial scan
        int cm = 0, cu = 0;
        for (int n = 0; n < N_; ++n) {
            if (maskf[n]) { list[cm++] = (unsigned short)n; }
            else          { zr[n] = (unsigned short)cu; ++cu; }
        }
        num_one = cm;                                    // >=1 (central pixel)
    }
    __syncthreads();

    const int no = num_one;
    for (int n = threadIdx.x; n < N_; n += blockDim.x)
        src[b * N_ + n] = maskf[n] ? n : (int)list[zr[n] % no];
}

// ---------------------------------------------------------------------------
// Kernel 2: weights -> padded bf16  Wp[oc][tap*224 + c]
// ---------------------------------------------------------------------------
__global__ void build_w_kernel(const float* __restrict__ W,
                               unsigned short* __restrict__ Wp) {
    const int idx = blockIdx.x * blockDim.x + threadIdx.x;
    if (idx >= OUTC_ * KTOT_) return;
    const int oc = idx / KTOT_, r = idx % KTOT_;
    const int tap = r / CP_,    c = r % CP_;
    const float v = (c < CH_) ? W[(oc * CH_ + c) * TAPS_ + tap] : 0.f;
    Wp[idx] = f2bf(v);
}

// ---------------------------------------------------------------------------
// Kernel 3: gather + convert:  Act[b][n][0..223] = bf16( x[b][c][src[b][n]] )
// one thread per (b,n); x rows for a batch stay L2-resident (192 MB L2)
// ---------------------------------------------------------------------------
__global__ void build_act_kernel(const float* __restrict__ x,
                                 const int* __restrict__ src,
                                 unsigned short* __restrict__ Act) {
    const int idx = blockIdx.x * blockDim.x + threadIdx.x;
    if (idx >= B_ * N_) return;
    const int b = idx / N_;
    const int s = src[idx];
    const float* xb = x + (size_t)b * C_ * N_ + s;       // gather column
    unsigned short* dst = Act + (size_t)idx * CP_;

    #pragma unroll 1
    for (int c0 = 0; c0 < CH_; c0 += 8) {
        unsigned short h[8];
        #pragma unroll
        for (int j = 0; j < 8; ++j) h[j] = f2bf(xb[(size_t)(c0 + j) * N_]);
        const unsigned int w0 = (unsigned int)h[0] | ((unsigned int)h[1] << 16);
        const unsigned int w1 = (unsigned int)h[2] | ((unsigned int)h[3] << 16);
        const unsigned int w2 = (unsigned int)h[4] | ((unsigned int)h[5] << 16);
        const unsigned int w3 = (unsigned int)h[6] | ((unsigned int)h[7] << 16);
        *(uint4*)(dst + c0) = make_uint4(w0, w1, w2, w3);
    }
    *(uint4*)(dst + 200) = make_uint4(0, 0, 0, 0);       // pad 200..223
    *(uint4*)(dst + 208) = make_uint4(0, 0, 0, 0);
    *(uint4*)(dst + 216) = make_uint4(0, 0, 0, 0);
}

// ---------------------------------------------------------------------------
// Kernel 4: implicit-im2col GEMM, bf16 WMMA 16x16x32, f32 accumulate.
// 4 wave32 = 128 thr; tile = 64 oc x 64 pixels; K = 2016 (63 chunks of 32).
// Double-buffered LDS fed by GLOBAL_LOAD_ASYNC_TO_LDS_B128 (ASYNCcnt).
// ---------------------------------------------------------------------------
__global__ __launch_bounds__(128)
void conv_wmma_kernel(const unsigned short* __restrict__ Act,
                      const unsigned short* __restrict__ Wp,
                      const float* __restrict__ bias,
                      const unsigned short* __restrict__ zero_pg,
                      float* __restrict__ out) {
    const int n0 = blockIdx.x * 64;       // pixel-tile base
    const int b  = blockIdx.y;

    __shared__ __align__(16) unsigned short w_lds[2][64 * 32];  // [buf][oc][k]
    __shared__ __align__(16) unsigned short a_lds[2][64 * 32];  // [buf][pix][k]

    const int t    = threadIdx.x;
    const int lane = t & 31;
    const int wv   = t >> 5;              // wave id 0..3 -> oc base 16*wv
    const int m    = lane & 15;
    const int hi   = lane >> 4;

    // ---- per-thread staging invariants (2 units of 16 B each) ----
    int      rowi[2], qi[2], py_[2], px_[2], nval[2];
    const unsigned short* wgp[2];         // weight global base (chunk 0)
    const unsigned short* agp[2];         // activation row base (tap-center)
    unsigned wlo[2], alo[2];              // LDS byte offsets (buffer 0)
    const unsigned wbase = (unsigned)(size_t)&w_lds[0][0];
    const unsigned abase = (unsigned)(size_t)&a_lds[0][0];
    const size_t   act_b = (size_t)b * N_ * CP_;
    #pragma unroll
    for (int rep = 0; rep < 2; ++rep) {
        const int i = t + rep * 128;      // 0..255
        rowi[rep] = i >> 2;               // oc (weights) / local pixel (acts)
        qi[rep]   = i & 3;
        wgp[rep]  = Wp + (size_t)rowi[rep] * KTOT_ + qi[rep] * 8;
        const int n = n0 + rowi[rep];
        nval[rep] = (n < N_);
        py_[rep]  = n / P_;
        px_[rep]  = n % P_;
        agp[rep]  = Act + act_b + (size_t)n * CP_ + qi[rep] * 8;
        wlo[rep]  = wbase + (unsigned)(rowi[rep] * 32 + qi[rep] * 8) * 2u;
        alo[rep]  = abase + (unsigned)(rowi[rep] * 32 + qi[rep] * 8) * 2u;
    }

    // issue async global->LDS copies for one K-chunk into buffer `buf`
    auto issue = [&](int chunk, int buf) {
        const int tap = chunk / 7;                       // uniform scalar math
        const int c0  = chunk * 32 - tap * CP_;
        const int dy  = tap / 3 - 1, dx = tap % 3 - 1;
        const int shift = (dy * P_ + dx) * CP_ + c0;     // halves
        const unsigned lbuf = (unsigned)buf * 4096u;     // LDS buffer stride
        #pragma unroll
        for (int rep = 0; rep < 2; ++rep) {
            // weights: offset within row is just 32*chunk halves
            const unsigned long long wg =
                (unsigned long long)(size_t)(wgp[rep] + chunk * 32);
            asm volatile("global_load_async_to_lds_b128 %0, %1, off"
                         :: "v"(wlo[rep] + lbuf), "v"(wg) : "memory");
            // activations: shifted row, zero page when out of bounds
            const int sy = py_[rep] + dy, sx = px_[rep] + dx;
            const bool ok = nval[rep] &&
                            ((unsigned)sy < (unsigned)P_) &&
                            ((unsigned)sx < (unsigned)P_);
            const unsigned short* ap = ok ? (agp[rep] + shift) : zero_pg;
            const unsigned long long ag = (unsigned long long)(size_t)ap;
            asm volatile("global_load_async_to_lds_b128 %0, %1, off"
                         :: "v"(alo[rep] + lbuf), "v"(ag) : "memory");
        }
    };

    v8f acc[4];
    #pragma unroll
    for (int s = 0; s < 4; ++s)
        #pragma unroll
        for (int r = 0; r < 8; ++r) acc[s][r] = 0.f;

    issue(0, 0);                                          // prologue

    for (int it = 0; it < CHUNKS_; ++it) {
        const int buf = it & 1;
        if (it + 1 < CHUNKS_) {
            issue(it + 1, buf ^ 1);                       // prefetch next chunk
            // 4 asyncs just issued remain; chunk `it`'s 4 must be complete
            asm volatile("s_wait_asynccnt 0x4" ::: "memory");
        } else {
            asm volatile("s_wait_asynccnt 0x0" ::: "memory");
        }
        __syncthreads();                                  // LDS visible to all

        // ---- fragments: contiguous k order (k = 16*hi + s) in both A and B ----
        union { uint4 q[2]; v16bf v; } af, bfu;
        {   // A (weights) 16x32: lanes<16 K0..15; lanes>=16 K16..31
            const unsigned short* ap = &w_lds[buf][(16 * wv + m) * 32];
            af.q[0] = *(const uint4*)(ap + hi * 16);
            af.q[1] = *(const uint4*)(ap + hi * 16 + 8);
        }
        #pragma unroll
        for (int s = 0; s < 4; ++s) {
            // B (acts) 32x16: lanes<16 K0..15, lanes>=16 K16..31; N = lane%16
            const unsigned short* bp = &a_lds[buf][(s * 16 + m) * 32 + hi * 16];
            bfu.q[0] = *(const uint4*)(bp);
            bfu.q[1] = *(const uint4*)(bp + 8);
            acc[s] = __builtin_amdgcn_wmma_f32_16x16x32_bf16(
                         false, af.v, false, bfu.v,
                         (short)0, acc[s], false, false);
        }
        __syncthreads();           // compute done before buffer is re-filled
    }

    // ---- store: D VGPR r -> M = r + 8*hi, N = lane%16 ----
    #pragma unroll
    for (int s = 0; s < 4; ++s) {
        const int n = n0 + s * 16 + m;
        if (n < N_) {
            #pragma unroll
            for (int r = 0; r < 8; ++r) {
                const int oc = 16 * wv + r + hi * 8;
                out[((size_t)b * OUTC_ + oc) * N_ + n] = acc[s][r] + bias[oc];
            }
        }
    }
}

// ---------------------------------------------------------------------------
extern "C" void kernel_launch(void* const* d_in, const int* in_sizes, int n_in,
                              void* d_out, int out_size, void* d_ws, size_t ws_size,
                              hipStream_t stream) {
    const float* x        = (const float*)d_in[0];  // (B,202,27,27)
    const float* W        = (const float*)d_in[1];  // (64,200,3,3)
    const float* bias     = (const float*)d_in[2];  // (64,)
    const int*   rand_idx = (const int*)d_in[3];    // (B,58)
    float*       out      = (float*)d_out;          // (B,64,27,27)

    char* ws = (char*)d_ws;
    const size_t ZP_BYTES  = 256;                                    // zero page
    const size_t SRC_BYTES = (size_t)B_ * N_ * sizeof(int);          // 1.49 MB
    const size_t WP_BYTES  = (size_t)OUTC_ * KTOT_ * sizeof(short);  // 258 KB
    unsigned int*   zp  = (unsigned int*)ws;
    int*            src = (int*)(ws + ZP_BYTES);
    unsigned short* Wp  = (unsigned short*)(ws + ZP_BYTES + SRC_BYTES);
    unsigned short* Act = (unsigned short*)(ws + ZP_BYTES + SRC_BYTES + WP_BYTES);

    // 0) zero page for async-copy padding
    init_zero_kernel<<<1, 64, 0, stream>>>(zp);

    // 1) EPF source indices (one block per batch image)
    build_src_kernel<<<B_, 256, 0, stream>>>(x, rand_idx, src);

    // 2) weights -> padded bf16
    {
        const int total = OUTC_ * KTOT_;
        build_w_kernel<<<(total + 255) / 256, 256, 0, stream>>>(W, Wp);
    }

    // 3) filtered activations -> bf16 [b][n][224]
    {
        const int total = B_ * N_;
        build_act_kernel<<<(total + 255) / 256, 256, 0, stream>>>(x, src, Act);
    }

    // 4) implicit GEMM conv: grid = (pixel tiles, batch)
    {
        dim3 grid((N_ + 63) / 64, B_);
        conv_wmma_kernel<<<grid, 128, 0, stream>>>(Act, Wp, bias,
                                                   (const unsigned short*)zp, out);
    }
}
